// RectConv2d_71897752535558
// MI455X (gfx1250) — compile-verified, matrix operand order checked by reading, not run on hardware
//
#include <hip/hip_runtime.h>
#include <math.h>

typedef __attribute__((ext_vector_type(16))) _Float16 v16h;
typedef __attribute__((ext_vector_type(8)))  float    v8f;

#define NB   4
#define NC   64
#define NOC  64
#define NH   80
#define NW   80
#define HW   (NH*NW)        // 6400
#define NPIX (NB*HW)        // 25600
#define TMAX 49
#define RCHMAX 98           // max K-chunks for rect GEMM: 49*64/32
#define CCH  18             // K-chunks for 3x3 convs: 576/32

#define SEL_NX  0
#define SEL_NY  1
#define SEL_NT  2
#define SEL_IDX 3
#define SEL_NCH 4

// ---------------------------------------------------------------------------
// WMMA fragment index helper (ISA 7.12.2, 16-bit A 16x32 / mirrored B 32x16):
// lane = {M or N}(low 4 bits) | halfK(bit4); element e -> K = (e<8?e:e+8)+8*half
__device__ __forceinline__ int frag_k(int lane, int e) {
    return ((e < 8) ? e : e + 8) + ((lane >> 4) << 3);
}

__device__ __forceinline__ float fetchpad(const float* __restrict__ xc, int ix, int iy) {
    // ix,iy are coordinates in the zero-padded (82x82) plane
    return (ix >= 1 && ix <= NH && iy >= 1 && iy <= NW) ? xc[(ix - 1) * NW + (iy - 1)] : 0.f;
}

// ---------------------------------------------------------------------------
__global__ void k_init(float* sums) {
    if (threadIdx.x < 2) sums[threadIdx.x] = 0.f;
}

// lm / wm / theta : three 64->1 3x3 convs + mean accumulation
__global__ void __launch_bounds__(256) k_lwt(
    const float* __restrict__ x,
    const float* __restrict__ lw, const float* __restrict__ lb,
    const float* __restrict__ ww, const float* __restrict__ wb,
    const float* __restrict__ tw, const float* __restrict__ tb,
    float* __restrict__ lm, float* __restrict__ wm, float* __restrict__ th,
    float* sums)
{
    __shared__ float s_lw[576], s_ww[576], s_tw[576];
    __shared__ float s_red[256];
    int tid = threadIdx.x;
    for (int i = tid; i < 576; i += 256) { s_lw[i] = lw[i]; s_ww[i] = ww[i]; s_tw[i] = tw[i]; }
    __syncthreads();

    int p = blockIdx.x * 256 + tid;              // NPIX == gridDim*256 exactly
    int b = p / HW, r = p % HW, h = r / NW, w = r % NW;
    float sl = 0.f, sw = 0.f, st = 0.f;
    const float* xb = x + (size_t)b * NC * HW;
    for (int c = 0; c < NC; ++c) {
        const float* xc = xb + c * HW;
        #pragma unroll
        for (int t = 0; t < 9; ++t) {
            int hy = h + t / 3 - 1, wx = w + t % 3 - 1;
            float v = (hy >= 0 && hy < NH && wx >= 0 && wx < NW) ? xc[hy * NW + wx] : 0.f;
            int wi = c * 9 + t;
            sl += v * s_lw[wi]; sw += v * s_ww[wi]; st += v * s_tw[wi];
        }
    }
    float lmv = 8.f / (1.f + expf(-(sl + lb[0]))) + 1.f;   // sigmoid*8+1
    float wmv = 8.f / (1.f + expf(-(sw + wb[0]))) + 1.f;
    lm[p] = lmv; wm[p] = wmv; th[p] = st + tb[0];

    s_red[tid] = lmv; __syncthreads();
    for (int s = 128; s > 0; s >>= 1) { if (tid < s) s_red[tid] += s_red[tid + s]; __syncthreads(); }
    if (tid == 0) atomicAdd(&sums[0], s_red[0]);
    __syncthreads();
    s_red[tid] = wmv; __syncthreads();
    for (int s = 128; s > 0; s >>= 1) { if (tid < s) s_red[tid] += s_red[tid + s]; __syncthreads(); }
    if (tid == 0) atomicAdd(&sums[1], s_red[0]);
}

// data-dependent kernel-size selection (device side; launches stay static)
__global__ void k_sel(const float* sums, int* sel) {
    float mL = sums[0] / (float)NPIX, mW = sums[1] / (float)NPIX;
    int nx = (int)floorf(mL); if ((nx & 1) == 0) nx -= 1; if (nx < 3) nx = 3; if (nx > 7) nx = 7;
    int ny = (int)floorf(mW); if ((ny & 1) == 0) ny -= 1; if (ny < 3) ny = 3; if (ny > 7) ny = 7;
    int idx, code = nx * 10 + ny;
    switch (code) {
        case 33: idx = 0; break; case 35: idx = 1; break; case 53: idx = 2; break;
        case 37: idx = 3; break; case 73: idx = 4; break; case 55: idx = 5; break;
        case 57: idx = 6; break; case 75: idx = 7; break; default: idx = 8; break;
    }
    sel[SEL_NX] = nx; sel[SEL_NY] = ny; sel[SEL_NT] = nx * ny;
    sel[SEL_IDX] = idx; sel[SEL_NCH] = nx * ny * 2;          // K/32 = NT*64/32
}

// pre-swizzle m_w / b_w into f16 WMMA-A fragment order, k = t*64 + c
__global__ void k_apre_conv(const float* __restrict__ mw, const float* __restrict__ bw,
                            unsigned short* apm, unsigned short* apb)
{
    int id = blockIdx.x * 256 + threadIdx.x;                 // 2*CCH*2048 threads exactly
    const float* src = (id < CCH * 2048) ? mw : bw;
    _Float16* dst = (_Float16*)((id < CCH * 2048) ? apm : apb);
    int r = id % (CCH * 2048);
    int chunk = r / 2048, rm = r % 2048;
    int mtile = rm / 512, q = rm % 512;
    int lane = q >> 4, e = q & 15, m = lane & 15;
    int k = chunk * 32 + frag_k(lane, e);
    int c = k & 63, t = k >> 6;
    dst[(size_t)(chunk * 4 + mtile) * 512 + q] = (_Float16)src[((mtile * 16 + m) * 64 + c) * 9 + t];
}

// pre-swizzle the SELECTED rect weight (one of 9) into fragment order
__global__ void k_apre_rect(const int* sel,
    const float* w0, const float* w1, const float* w2, const float* w3, const float* w4,
    const float* w5, const float* w6, const float* w7, const float* w8,
    unsigned short* apr)
{
    int id = blockIdx.x * 256 + threadIdx.x;                 // RCHMAX*2048 threads exactly
    int NT = sel[SEL_NT], K = NT * 64, idx = sel[SEL_IDX];
    const float* src;
    switch (idx) {
        case 0: src = w0; break; case 1: src = w1; break; case 2: src = w2; break;
        case 3: src = w3; break; case 4: src = w4; break; case 5: src = w5; break;
        case 6: src = w6; break; case 7: src = w7; break; default: src = w8; break;
    }
    int chunk = id / 2048, rm = id % 2048;
    int mtile = rm / 512, q = rm % 512;
    int lane = q >> 4, e = q & 15, m = lane & 15;
    int k = chunk * 32 + frag_k(lane, e);
    float v = 0.f;
    if (k < K) { int c = k & 63, t = k >> 6; v = src[((mtile * 16 + m) * 64 + c) * NT + t]; }
    ((_Float16*)apr)[(size_t)(chunk * 4 + mtile) * 512 + q] = (_Float16)v;
}

// m = tanh(conv3x3) and bias = conv3x3 : WMMA implicit GEMM, K = 9*64 = 576
// 8 waves: waves 0-3 -> m (4 M-tiles), waves 4-7 -> bias; shared im2col B tile in LDS
__global__ void __launch_bounds__(256) k_conv_mb(
    const float* __restrict__ x,
    const unsigned short* __restrict__ apm, const unsigned short* __restrict__ apb,
    const float* __restrict__ mb, const float* __restrict__ bb,
    float* __restrict__ mArr, float* __restrict__ bArr)
{
    __shared__ __align__(32) _Float16 bT[512];
    int tid = threadIdx.x, lane = tid & 31, wave = tid >> 5;
    int pixBase = blockIdx.x * 16;

    int pj[2], bj[2], hj[2], wj[2], klj[2];
    #pragma unroll
    for (int j = 0; j < 2; ++j) {
        int idx = tid * 2 + j, lf = idx >> 4, e = idx & 15;
        klj[j] = frag_k(lf, e);
        int p = pixBase + (lf & 15); pj[j] = p;
        bj[j] = p / HW; int r = p % HW; hj[j] = r / NW; wj[j] = r % NW;
    }
    int mtile = wave & 3;
    bool isM = wave < 4;
    const _Float16* Af = (const _Float16*)(isM ? apm : apb);
    v8f acc = {};
    for (int chunk = 0; chunk < CCH; ++chunk) {
        #pragma unroll
        for (int j = 0; j < 2; ++j) {
            int k = chunk * 32 + klj[j];
            int c = k & 63, t = k >> 6;                      // k = t*64 + c
            int hy = hj[j] + t / 3 - 1, wx = wj[j] + t % 3 - 1;
            float v = (hy >= 0 && hy < NH && wx >= 0 && wx < NW)
                      ? x[((size_t)(bj[j] * NC + c)) * HW + hy * NW + wx] : 0.f;
            bT[tid * 2 + j] = (_Float16)v;
        }
        __syncthreads();
        v16h bf = *(const v16h*)(&bT[lane * 16]);
        v16h af = *(const v16h*)(Af + (size_t)(chunk * 4 + mtile) * 512 + lane * 16);
        acc = __builtin_amdgcn_wmma_f32_16x16x32_f16(false, af, false, bf, (short)0, acc, false, false);
        __syncthreads();
    }
    #pragma unroll
    for (int r = 0; r < 8; ++r) {
        int o = mtile * 16 + r + ((lane >> 4) << 3);         // C/D layout: VGPR r -> M = r / r+8
        int p = pixBase + (lane & 15);
        int b = p / HW, rr = p % HW;
        size_t oi = ((size_t)(b * NOC + o)) * HW + rr;
        if (isM) mArr[oi] = tanhf(acc[r] + mb[o]);
        else     bArr[oi] = acc[r] + bb[o];
    }
}

// bilinear sampling metadata per (pixel, tap): 4 weights + 4 packed u8 indices
__global__ void __launch_bounds__(256) k_meta(
    const int* sel,
    const float* __restrict__ lm, const float* __restrict__ wm, const float* __restrict__ th,
    float4* __restrict__ mw4, unsigned int* __restrict__ mi)
{
    int id = blockIdx.x * 256 + threadIdx.x;                 // NPIX*TMAX exactly
    int p = id / TMAX, t = id % TMAX;
    int NX = sel[SEL_NX], NY = sel[SEL_NY], NT = sel[SEL_NT];
    if (t >= NT) return;
    int r = p % HW, h = r / NW, w = r % NW;
    float theta = th[p], lmv = lm[p], wmv = wm[p];
    int tx = t / NY, ty = t % NY;
    float pnx = (float)(tx - ((NX - 1) >> 1));
    float pny = (float)(ty - ((NY - 1) >> 1));
    float ct = cosf(theta), st = sinf(theta);
    float px = (float)(h + 1) + (lmv / (float)NX) * (pnx * ct - pny * st);
    float py = (float)(w + 1) + (wmv / (float)NY) * (pnx * st + pny * ct);
    float qltx = fminf(fmaxf(floorf(px), 0.f), 81.f);
    float qlty = fminf(fmaxf(floorf(py), 0.f), 81.f);
    float qrbx = fminf(fmaxf(floorf(px) + 1.f, 0.f), 81.f);
    float qrby = fminf(fmaxf(floorf(py) + 1.f, 0.f), 81.f);
    float pxc = fminf(fmaxf(px, 0.f), 81.f);
    float pyc = fminf(fmaxf(py, 0.f), 81.f);
    float g_lt = (1.f + (qltx - pxc)) * (1.f + (qlty - pyc));
    float g_rb = (1.f - (qrbx - pxc)) * (1.f - (qrby - pyc));
    float g_lb = (1.f + (qltx - pxc)) * (1.f - (qrby - pyc));
    float g_rt = (1.f - (qrbx - pxc)) * (1.f + (qlty - pyc));
    mw4[id] = make_float4(g_lt, g_rb, g_lb, g_rt);
    mi[id] = (unsigned)qltx | ((unsigned)qlty << 8) | ((unsigned)qrbx << 16) | ((unsigned)qrby << 24);
}

// rect GEMM: out[o,p] = m * (sum_k rect_w * x_off + rect_b) + bias ; K = NT*64
__global__ void __launch_bounds__(128) k_rect(
    const float* __restrict__ x,
    const unsigned short* __restrict__ apr,
    const float4* __restrict__ mw4, const unsigned int* __restrict__ mi,
    const float* __restrict__ mArr, const float* __restrict__ bArr,
    const int* sel,
    const float* rb0, const float* rb1, const float* rb2, const float* rb3, const float* rb4,
    const float* rb5, const float* rb6, const float* rb7, const float* rb8,
    float* __restrict__ out)
{
    __shared__ __align__(32) _Float16 bT[512];
    int tid = threadIdx.x, lane = tid & 31, mtile = tid >> 5;
    int pixBase = blockIdx.x * 16;

    int pj[4], bj[4], klj[4];
    #pragma unroll
    for (int j = 0; j < 4; ++j) {
        int idx = tid * 4 + j, lf = idx >> 4, e = idx & 15;
        klj[j] = frag_k(lf, e);
        int p = pixBase + (lf & 15); pj[j] = p; bj[j] = p / HW;
    }
    int nCh = __builtin_amdgcn_readfirstlane(sel[SEL_NCH]); // scalar loop bound -> EXEC stays all-1s
    int idx = __builtin_amdgcn_readfirstlane(sel[SEL_IDX]);
    const float* rb;
    switch (idx) {
        case 0: rb = rb0; break; case 1: rb = rb1; break; case 2: rb = rb2; break;
        case 3: rb = rb3; break; case 4: rb = rb4; break; case 5: rb = rb5; break;
        case 6: rb = rb6; break; case 7: rb = rb7; break; default: rb = rb8; break;
    }
    const _Float16* Af = (const _Float16*)apr;
    v8f acc = {};
    for (int chunk = 0; chunk < nCh; ++chunk) {
        #pragma unroll
        for (int j = 0; j < 4; ++j) {
            int k = chunk * 32 + klj[j];                     // k = t*64 + c, always < K
            int c = k & 63, t = k >> 6;
            float4 g = mw4[(size_t)pj[j] * TMAX + t];
            unsigned int pk = mi[(size_t)pj[j] * TMAX + t];
            int ax = pk & 255, ay = (pk >> 8) & 255, bx = (pk >> 16) & 255, by = pk >> 24;
            const float* xc = x + ((size_t)(bj[j] * NC + c)) * HW;
            float v = g.x * fetchpad(xc, ax, ay) + g.y * fetchpad(xc, bx, by)
                    + g.z * fetchpad(xc, ax, by) + g.w * fetchpad(xc, bx, ay);
            bT[tid * 4 + j] = (_Float16)v;
        }
        __syncthreads();
        v16h bf = *(const v16h*)(&bT[lane * 16]);
        v16h af = *(const v16h*)(Af + (size_t)(chunk * 4 + mtile) * 512 + lane * 16);
        __builtin_prefetch(Af + (size_t)((chunk + 1) * 4 + mtile) * 512 + lane * 16, 0, 1);
        acc = __builtin_amdgcn_wmma_f32_16x16x32_f16(false, af, false, bf, (short)0, acc, false, false);
        __syncthreads();
    }
    #pragma unroll
    for (int r = 0; r < 8; ++r) {
        int o = mtile * 16 + r + ((lane >> 4) << 3);
        int p = pixBase + (lane & 15);
        int b = p / HW, rr = p % HW;
        size_t oi = ((size_t)(b * NOC + o)) * HW + rr;
        out[oi] = mArr[oi] * (acc[r] + rb[o]) + bArr[oi];
    }
}

// ---------------------------------------------------------------------------
extern "C" void kernel_launch(void* const* d_in, const int* in_sizes, int n_in,
                              void* d_out, int out_size, void* d_ws, size_t ws_size,
                              hipStream_t stream)
{
    (void)in_sizes; (void)n_in; (void)out_size; (void)ws_size;
    const float* x    = (const float*)d_in[0];
    const float* m_w  = (const float*)d_in[5];
    const float* m_b  = (const float*)d_in[6];
    const float* b_w  = (const float*)d_in[7];
    const float* b_b  = (const float*)d_in[8];
    const float* l_w  = (const float*)d_in[9];
    const float* l_b  = (const float*)d_in[10];
    const float* w_w  = (const float*)d_in[11];
    const float* w_b  = (const float*)d_in[12];
    const float* th_w = (const float*)d_in[13];
    const float* th_b = (const float*)d_in[14];
    const float* rw[9]; const float* rb[9];
    for (int i = 0; i < 9; ++i) { rw[i] = (const float*)d_in[15 + 2 * i]; rb[i] = (const float*)d_in[16 + 2 * i]; }

    char* ws = (char*)d_ws;
    size_t off = 0;
    auto take = [&](size_t bytes) -> char* {
        char* pp = ws + off; off = (off + bytes + 255) & ~(size_t)255; return pp;
    };
    float*          sums = (float*)take(2 * sizeof(float));
    int*            sel  = (int*)take(8 * sizeof(int));
    float*          lm   = (float*)take((size_t)NPIX * 4);
    float*          wm   = (float*)take((size_t)NPIX * 4);
    float*          th   = (float*)take((size_t)NPIX * 4);
    float*          mArr = (float*)take((size_t)NPIX * NOC * 4);
    float*          bArr = (float*)take((size_t)NPIX * NOC * 4);
    float4*         mw4  = (float4*)take((size_t)NPIX * TMAX * 16);
    unsigned int*   mi   = (unsigned int*)take((size_t)NPIX * TMAX * 4);
    unsigned short* apr  = (unsigned short*)take((size_t)RCHMAX * 2048 * 2);
    unsigned short* apm  = (unsigned short*)take((size_t)CCH * 2048 * 2);
    unsigned short* apb  = (unsigned short*)take((size_t)CCH * 2048 * 2);

    k_init<<<1, 32, 0, stream>>>(sums);
    k_lwt<<<NPIX / 256, 256, 0, stream>>>(x, l_w, l_b, w_w, w_b, th_w, th_b, lm, wm, th, sums);
    k_sel<<<1, 1, 0, stream>>>(sums, sel);
    k_apre_conv<<<(2 * CCH * 2048) / 256, 256, 0, stream>>>(m_w, b_w, apm, apb);
    k_apre_rect<<<(RCHMAX * 2048) / 256, 256, 0, stream>>>(sel,
        rw[0], rw[1], rw[2], rw[3], rw[4], rw[5], rw[6], rw[7], rw[8], apr);
    k_conv_mb<<<NPIX / 16, 256, 0, stream>>>(x, apm, apb, m_b, b_b, mArr, bArr);
    k_meta<<<(NPIX * TMAX) / 256, 256, 0, stream>>>(sel, lm, wm, th, mw4, mi);
    k_rect<<<NPIX / 16, 128, 0, stream>>>(x, apr, mw4, mi, mArr, bArr, sel,
        rb[0], rb[1], rb[2], rb[3], rb[4], rb[5], rb[6], rb[7], rb[8], (float*)d_out);
}
